// SummaryAttention_14061722927961
// MI455X (gfx1250) — matrix operand order repaired and checked by reading, not compiled, not run on hardware
//
#include <hip/hip_runtime.h>

// ---------------------------------------------------------------------------
// Summary attention for MI455X (gfx1250, wave32, WMMA).
// All matmuls on v_wmma_f32_16x16x32_bf16 (f32 accumulate). Operands are
// pre-converted to bf16; boolean masks are pre-packed to bitmask words.
// Staging uses global_load_async_to_lds_b128 when available.
// ---------------------------------------------------------------------------

#define E_DIM 1024
#define HEADS 16
#define HDIM  64
#define B_DIM 2
#define S_LEN 256
#define R_LEN 2048
#define CH    64          // attention key-chunk size

typedef __attribute__((ext_vector_type(4)))  int    v4i;
typedef __attribute__((ext_vector_type(4)))  __bf16 v4bf;
typedef __attribute__((ext_vector_type(8)))  __bf16 v8bf;
typedef __attribute__((ext_vector_type(16))) __bf16 v16bf;
typedef __attribute__((ext_vector_type(8)))  float  v8f;

#if defined(__has_builtin)
#if __has_builtin(__builtin_amdgcn_global_load_async_to_lds_b128) && \
    __has_builtin(__builtin_amdgcn_s_wait_asynccnt)
#define USE_ASYNC_LDS 1
#endif
#endif
#ifndef USE_ASYNC_LDS
#define USE_ASYNC_LDS 0
#endif

// f32 -> bf16 round-to-nearest-even via bit ops
__device__ __forceinline__ __bf16 f2bf(float f) {
    unsigned u = __builtin_bit_cast(unsigned, f);
    unsigned r = (u + 0x7FFFu + ((u >> 16) & 1u)) >> 16;
    unsigned short s = (unsigned short)r;
    return __builtin_bit_cast(__bf16, s);
}

// Copy 16 bytes (8 bf16) global -> LDS.
__device__ __forceinline__ void stage16B(const __bf16* g, __bf16* l) {
#if USE_ASYNC_LDS
    __builtin_amdgcn_global_load_async_to_lds_b128(
        (__attribute__((address_space(1))) v4i*)(v4i*)g,
        (__attribute__((address_space(3))) v4i*)(v4i*)l, 0, 0);
#else
    *(v8bf*)l = *(const v8bf*)g;
#endif
}
__device__ __forceinline__ void stage_wait() {
#if USE_ASYNC_LDS
    __builtin_amdgcn_s_wait_asynccnt(0);
#endif
}

// Build a 16-element bf16 vector from two 16B-aligned runs of 8 in LDS.
__device__ __forceinline__ v16bf pack16(const __bf16* p0, const __bf16* p1) {
    v8bf lo = *(const v8bf*)p0;
    v8bf hi = *(const v8bf*)p1;
    v16bf r;
#pragma unroll
    for (int i = 0; i < 8; ++i) { r[i] = lo[i]; r[8 + i] = hi[i]; }
    return r;
}

__device__ __forceinline__ v8f wmma_bf16(v16bf a, v16bf b, v8f c) {
    return __builtin_amdgcn_wmma_f32_16x16x32_bf16(false, a, false, b,
                                                   (short)0, c, false, false);
}

// ---------------------------------------------------------------------------
// Elementwise f32 -> bf16 (n multiple of 4).
// ---------------------------------------------------------------------------
__global__ void __launch_bounds__(256)
cvt_bf16(const float* __restrict__ s, __bf16* __restrict__ d, int n) {
    int i = (blockIdx.x * 256 + threadIdx.x) * 4;
    if (i >= n) return;
    float4 v = *(const float4*)(s + i);
    v4bf o;
    o[0] = f2bf(v.x); o[1] = f2bf(v.y); o[2] = f2bf(v.z); o[3] = f2bf(v.w);
    *(v4bf*)(d + i) = o;
}

// ---------------------------------------------------------------------------
// Pack a byte-bool mask into 32-bit words (bit i = byte i nonzero).
// ---------------------------------------------------------------------------
__global__ void __launch_bounds__(256)
pack_mask(const unsigned char* __restrict__ src, unsigned* __restrict__ dst,
          int nwords) {
    int i = blockIdx.x * 256 + threadIdx.x;
    if (i >= nwords) return;
    const unsigned* s = (const unsigned*)src + (size_t)i * 8;
    unsigned bits = 0;
#pragma unroll
    for (int w = 0; w < 8; ++w) {
        unsigned v = s[w];
        bits |= ((v        & 1u) << (w * 4 + 0));
        bits |= (((v >> 8)  & 1u) << (w * 4 + 1));
        bits |= (((v >> 16) & 1u) << (w * 4 + 2));
        bits |= (((v >> 24) & 1u) << (w * 4 + 3));
    }
    dst[i] = bits;
}

// ---------------------------------------------------------------------------
// Y[M,E] = X[M,E] @ W[E,E]^T + bias, X/W bf16.
// OUT_MODE: 0 = f32 row-major, 1 = bf16 row-major, 2 = bf16 (B,H,D,Ltok).
// Block: 256 threads / 8 waves, tile M=64 x N=128; wave (wr 0..3, wc 0..1)
// computes 16x64 = four 16x16 WMMA accumulators. K-step = 32.
// ---------------------------------------------------------------------------
template <int OUT_MODE>
__global__ void __launch_bounds__(256)
gemm_bias(const __bf16* __restrict__ X, const __bf16* __restrict__ W,
          const float* __restrict__ bias, void* __restrict__ Yv, int Ltok) {
    __shared__ __bf16 Xs[64][40];
    __shared__ __bf16 Ws[128][40];

    const int tid  = threadIdx.x;
    const int wave = tid >> 5, lane = tid & 31;
    const int mBase = blockIdx.y * 64;
    const int nBase = blockIdx.x * 128;
    const int wr = wave >> 1;
    const int wc = wave & 1;
    const int c0 = lane & 15, half = lane >> 4;

    v8f acc[4] = {{}, {}, {}, {}};

    for (int k0 = 0; k0 < E_DIM; k0 += 32) {
        __syncthreads();
        {
            int r = tid >> 2, c = (tid & 3) * 8;
            stage16B(X + (size_t)(mBase + r) * E_DIM + k0 + c, &Xs[r][c]);
        }
#pragma unroll
        for (int v = tid; v < 512; v += 256) {
            int r = v >> 2, c = (v & 3) * 8;
            stage16B(W + (size_t)(nBase + r) * E_DIM + k0 + c, &Ws[r][c]);
        }
        stage_wait();
        __syncthreads();

        const int am = wr * 16 + c0;
        const int kh = half * 8;
        v16bf a = pack16(&Xs[am][kh], &Xs[am][kh + 16]);
        const int bk = half * 8;
#pragma unroll
        for (int sub = 0; sub < 4; ++sub) {
            const int bn = wc * 64 + sub * 16 + c0;
            v16bf b = pack16(&Ws[bn][bk], &Ws[bn][bk + 16]);
            acc[sub] = wmma_bf16(a, b, acc[sub]);
        }
    }

    const int row = mBase + wr * 16 + half * 8;
#pragma unroll
    for (int sub = 0; sub < 4; ++sub) {
        const int col = nBase + wc * 64 + sub * 16 + c0;
        const float bv = bias[col];
#pragma unroll
        for (int j = 0; j < 8; ++j) {
            const float y = acc[sub][j] + bv;
            const int r = row + j;
            if constexpr (OUT_MODE == 0) {
                ((float*)Yv)[(size_t)r * E_DIM + col] = y;
            } else if constexpr (OUT_MODE == 1) {
                ((__bf16*)Yv)[(size_t)r * E_DIM + col] = f2bf(y);
            } else {
                const int bb = r / Ltok, l = r - bb * Ltok;
                const int h = col >> 6, d = col & 63;
                ((__bf16*)Yv)[(((size_t)bb * HEADS + h) * HDIM + d) * Ltok + l] =
                    f2bf(y);
            }
        }
    }
}

// ---------------------------------------------------------------------------
// RoPE in place on bf16 q and k. inv_freq[i] = 10000^(-i/32).
// ---------------------------------------------------------------------------
__global__ void __launch_bounds__(256)
rope_qk(__bf16* __restrict__ q, __bf16* __restrict__ k, int L, int posOff) {
    const int idx = blockIdx.x * 256 + threadIdx.x;
    const int i = idx & 31;
    const int h = (idx >> 5) & 15;
    const int rest = idx >> 9;
    const int l = rest % L;
    const int b = rest / L;
    if (b >= B_DIM) return;

    const float ang = (float)(l + posOff) *
                      __expf(-(float)i * (9.210340371976184f / 32.0f));
    const float c = __cosf(ang), s = __sinf(ang);

    const size_t base = ((size_t)b * L + l) * E_DIM + h * HDIM + i;
    float q1 = (float)q[base], q2 = (float)q[base + 32];
    q[base]      = f2bf(q1 * c - q2 * s);
    q[base + 32] = f2bf(q2 * c + q1 * s);
    float k1 = (float)k[base], k2 = (float)k[base + 32];
    k[base]      = f2bf(k1 * c - k2 * s);
    k[base + 32] = f2bf(k2 * c + k1 * s);
}

// ---------------------------------------------------------------------------
// Flash attention over concatenated sources [K0(L0) ; K1(L1)].
// Q/K: bf16 (B,L,E). V: bf16 transposed (B,H,D,Lsrc). Masks: bit-packed
// words (Lsrc/32 per row). Out: bf16 (B,Lq,E).
// Block = 128 query rows of one (b,h); 8 waves x 16 rows; key chunks of 64.
// Row sums come from an extra WMMA against an all-ones B matrix.
// ---------------------------------------------------------------------------
__global__ void __launch_bounds__(256)
attn(const __bf16* __restrict__ Q,
     const __bf16* __restrict__ K0, const __bf16* __restrict__ V0T, int L0,
     const __bf16* __restrict__ K1, const __bf16* __restrict__ V1T, int L1,
     const unsigned* __restrict__ MW0, const unsigned* __restrict__ MW1,
     __bf16* __restrict__ Out, int Lq) {
    __shared__ __bf16 Qs[128][64];
    __shared__ __bf16 Ks[64][64];      // [key][d]
    __shared__ __bf16 Vt[64][64];      // [d][key]
    __shared__ __bf16 Ps[8][16][64];

    const int b = blockIdx.z, h = blockIdx.y;
    const int qBase = blockIdx.x * 128;
    const int tid = threadIdx.x;
    const int wave = tid >> 5, lane = tid & 31;
    const int c0 = lane & 15, half = lane >> 4;
    const int kh = half * 8;
    const float scale = 0.125f;        // D^-0.5

    const __bf16* qph = Q + (size_t)b * Lq * E_DIM + h * HDIM;
#pragma unroll
    for (int v = tid; v < 1024; v += 256) {
        int r = v >> 3, c = (v & 7) * 8;
        stage16B(qph + (size_t)(qBase + r) * E_DIM + c, &Qs[r][c]);
    }
    stage_wait();
    __syncthreads();

    const int am = wave * 16 + c0;
    v16bf aQ0 = pack16(&Qs[am][kh],      &Qs[am][kh + 16]);
    v16bf aQ1 = pack16(&Qs[am][32 + kh], &Qs[am][32 + kh + 16]);

    v16bf vones;
    {
        const __bf16 one = f2bf(1.0f);
#pragma unroll
        for (int i = 0; i < 16; ++i) vones[i] = one;
    }

    v8f o[4] = {{}, {}, {}, {}};
    float mrow[8], lrow[8];
#pragma unroll
    for (int j = 0; j < 8; ++j) { mrow[j] = -1e30f; lrow[j] = 0.0f; }

    const int Ltot = L0 + L1;
    for (int kc = 0; kc < Ltot; kc += CH) {
        const bool first = (kc < L0);
        const __bf16* Ksrc  = first ? K0 : K1;
        const __bf16* VsrcT = first ? V0T : V1T;
        const int Lsrc = first ? L0 : L1;
        const int kloc = first ? kc : (kc - L0);

        const __bf16* kph = Ksrc + (size_t)b * Lsrc * E_DIM + h * HDIM;
        const __bf16* vph = VsrcT + ((size_t)b * HEADS + h) * HDIM * Lsrc;

        __syncthreads();
#pragma unroll
        for (int v = tid; v < 512; v += 256) {   // K: 64x64 bf16
            int r = v >> 3, c = (v & 7) * 8;
            stage16B(kph + (size_t)(kloc + r) * E_DIM + c, &Ks[r][c]);
        }
#pragma unroll
        for (int v = tid; v < 512; v += 256) {   // V (transposed): 64x64 bf16
            int d = v >> 3, kk = (v & 7) * 8;
            stage16B(vph + (size_t)d * Lsrc + kloc + kk, &Vt[d][kk]);
        }
        if (kc + CH < Ltot) {                    // prefetch next K rows
            const bool nf = (kc + CH < L0);
            const __bf16* nkp = nf ? K0 : K1;
            const int nL = nf ? L0 : L1;
            const int nk = nf ? (kc + CH) : (kc + CH - L0);
            __builtin_prefetch(nkp + (size_t)b * nL * E_DIM + h * HDIM +
                                   (size_t)(nk + (tid >> 2)) * E_DIM, 0, 1);
        }
        stage_wait();
        __syncthreads();

        // ----- scores: 16 q-rows x 64 keys (4 key subtiles) -----
        v8f s[4] = {{}, {}, {}, {}};
#pragma unroll
        for (int t = 0; t < 4; ++t) {
            const int kr = t * 16 + c0;
            v16bf b0 = pack16(&Ks[kr][kh],      &Ks[kr][kh + 16]);
            s[t] = wmma_bf16(aQ0, b0, s[t]);
            v16bf b1 = pack16(&Ks[kr][32 + kh], &Ks[kr][32 + kh + 16]);
            s[t] = wmma_bf16(aQ1, b1, s[t]);
        }

        // ----- mask (bit-packed) + online softmax -----
        const unsigned* MW = first ? MW0 : MW1;
        const int wpr = Lsrc >> 5;               // mask words per row
        const int woff = kloc >> 5;
#pragma unroll
        for (int j = 0; j < 8; ++j) {
            const int grow = qBase + wave * 16 + half * 8 + j;
            uint2 mw = *(const uint2*)(MW + ((size_t)b * Lq + grow) * wpr + woff);
            float sv0 = ((mw.x >> c0)        & 1u) ? s[0][j] * scale : -1e30f;
            float sv1 = ((mw.x >> (c0 + 16)) & 1u) ? s[1][j] * scale : -1e30f;
            float sv2 = ((mw.y >> c0)        & 1u) ? s[2][j] * scale : -1e30f;
            float sv3 = ((mw.y >> (c0 + 16)) & 1u) ? s[3][j] * scale : -1e30f;

            float mx = fmaxf(fmaxf(sv0, sv1), fmaxf(sv2, sv3));
            mx = fmaxf(mx, __shfl_xor(mx, 1, 32));
            mx = fmaxf(mx, __shfl_xor(mx, 2, 32));
            mx = fmaxf(mx, __shfl_xor(mx, 4, 32));
            mx = fmaxf(mx, __shfl_xor(mx, 8, 32));

            const float mn   = fmaxf(mrow[j], mx);
            const float corr = __expf(mrow[j] - mn);
            mrow[j] = mn;
            lrow[j] *= corr;                     // row sum added via ones-WMMA
            o[0][j] *= corr; o[1][j] *= corr; o[2][j] *= corr; o[3][j] *= corr;

            const int pr = half * 8 + j;
            Ps[wave][pr][c0]      = f2bf(__expf(sv0 - mn));
            Ps[wave][pr][c0 + 16] = f2bf(__expf(sv1 - mn));
            Ps[wave][pr][c0 + 32] = f2bf(__expf(sv2 - mn));
            Ps[wave][pr][c0 + 48] = f2bf(__expf(sv3 - mn));
        }
        __syncthreads();   // Ps write -> read ordering (uniform control flow)

        v16bf aP0 = pack16(&Ps[wave][c0][kh],      &Ps[wave][c0][kh + 16]);
        v16bf aP1 = pack16(&Ps[wave][c0][32 + kh], &Ps[wave][c0][32 + kh + 16]);

        // ----- row sums: P @ ones -----
        v8f rs = {};
        rs = wmma_bf16(aP0, vones, rs);
        rs = wmma_bf16(aP1, vones, rs);
#pragma unroll
        for (int j = 0; j < 8; ++j) lrow[j] += rs[j];

        // ----- O += P @ V -----
#pragma unroll
        for (int n = 0; n < 4; ++n) {
            const int vr = n * 16 + c0;
            v16bf bv0 = pack16(&Vt[vr][kh],      &Vt[vr][kh + 16]);
            o[n] = wmma_bf16(aP0, bv0, o[n]);
            v16bf bv1 = pack16(&Vt[vr][32 + kh], &Vt[vr][32 + kh + 16]);
            o[n] = wmma_bf16(aP1, bv1, o[n]);
        }
    }

    // ----- epilogue: O / l (bf16 out) -----
    __bf16* oph = Out + (size_t)b * Lq * E_DIM + h * HDIM;
#pragma unroll
    for (int j = 0; j < 8; ++j) {
        const float inv = 1.0f / lrow[j];
        const size_t ro = (size_t)(qBase + wave * 16 + half * 8 + j) * E_DIM;
        oph[ro + 0  + c0] = f2bf(o[0][j] * inv);
        oph[ro + 16 + c0] = f2bf(o[1][j] * inv);
        oph[ro + 32 + c0] = f2bf(o[2][j] * inv);
        oph[ro + 48 + c0] = f2bf(o[3][j] * inv);
    }
}

// ---------------------------------------------------------------------------
extern "C" void kernel_launch(void* const* d_in, const int* in_sizes, int n_in,
                              void* d_out, int out_size, void* d_ws, size_t ws_size,
                              hipStream_t stream) {
    (void)in_sizes; (void)n_in; (void)out_size; (void)ws_size;

    const float* sum_x = (const float*)d_in[0];
    const float* reg_x = (const float*)d_in[1];
    const unsigned char* ss = (const unsigned char*)d_in[2];
    const unsigned char* sr = (const unsigned char*)d_in[3];
    const unsigned char* rs = (const unsigned char*)d_in[4];
    const unsigned char* rr = (const unsigned char*)d_in[5];

    const float* Wsrc[10];
    const float* bsrc[10];
    for (int i = 0; i < 10; ++i) {
        Wsrc[i] = (const float*)d_in[6 + 2 * i];
        bsrc[i] = (const float*)d_in[7 + 2 * i];
    }

    const size_t SE = (size_t)B_DIM * S_LEN * E_DIM;   // 524288
    const size_t RE = (size_t)B_DIM * R_LEN * E_DIM;   // 4194304
    const size_t EE = (size_t)E_DIM * E_DIM;           // 1048576

    __bf16* p = (__bf16*)d_ws;
    __bf16* wbf[10];
    for (int i = 0; i < 10; ++i) { wbf[i] = p; p += EE; }
    __bf16* xs_bf    = p; p += SE;
    __bf16* xr_bf    = p; p += RE;
    __bf16* sum_q    = p; p += SE;
    __bf16* sum_k    = p; p += SE;
    __bf16* sum_vT   = p; p += SE;
    __bf16* reg_q    = p; p += RE;
    __bf16* reg_k    = p; p += RE;
    __bf16* reg_vT   = p; p += RE;
    __bf16* sum_attn = p; p += SE;
    __bf16* k2_bf    = p; p += SE;
    __bf16* v2_T     = p; p += SE;
    __bf16* reg_attn = p; p += RE;

    // Bit-packed mask words
    const int nw_ss = B_DIM * S_LEN * (S_LEN / 32);    // 4096
    const int nw_sr = B_DIM * S_LEN * (R_LEN / 32);    // 32768
    const int nw_rs = B_DIM * R_LEN * (S_LEN / 32);    // 32768
    const int nw_rr = B_DIM * R_LEN * (R_LEN / 32);    // 262144
    unsigned* ssw = (unsigned*)p;
    unsigned* srw = ssw + nw_ss;
    unsigned* rsw = srw + nw_sr;
    unsigned* rrw = rsw + nw_rs;

    float* out_sum = (float*)d_out;
    float* out_reg = out_sum + SE;

    dim3 blk(256);
    dim3 gS(E_DIM / 128, (B_DIM * S_LEN) / 64);    // (8, 8)
    dim3 gR(E_DIM / 128, (B_DIM * R_LEN) / 64);    // (8, 64)

    // ---- bf16 conversions + mask packing ----
    for (int i = 0; i < 10; ++i)
        cvt_bf16<<<EE / 1024, blk, 0, stream>>>(Wsrc[i], wbf[i], (int)EE);
    cvt_bf16<<<SE / 1024, blk, 0, stream>>>(sum_x, xs_bf, (int)SE);
    cvt_bf16<<<RE / 1024, blk, 0, stream>>>(reg_x, xr_bf, (int)RE);
    pack_mask<<<(nw_ss + 255) / 256, blk, 0, stream>>>(ss, ssw, nw_ss);
    pack_mask<<<(nw_sr + 255) / 256, blk, 0, stream>>>(sr, srw, nw_sr);
    pack_mask<<<(nw_rs + 255) / 256, blk, 0, stream>>>(rs, rsw, nw_rs);
    pack_mask<<<(nw_rr + 255) / 256, blk, 0, stream>>>(rr, rrw, nw_rr);

    // ---- projections ----
    gemm_bias<1><<<gS, blk, 0, stream>>>(xs_bf, wbf[0], bsrc[0], sum_q,  S_LEN);
    gemm_bias<1><<<gS, blk, 0, stream>>>(xs_bf, wbf[1], bsrc[1], sum_k,  S_LEN);
    gemm_bias<2><<<gS, blk, 0, stream>>>(xs_bf, wbf[2], bsrc[2], sum_vT, S_LEN);
    gemm_bias<1><<<gR, blk, 0, stream>>>(xr_bf, wbf[4], bsrc[4], reg_q,  R_LEN);
    gemm_bias<1><<<gR, blk, 0, stream>>>(xr_bf, wbf[5], bsrc[5], reg_k,  R_LEN);
    gemm_bias<2><<<gR, blk, 0, stream>>>(xr_bf, wbf[6], bsrc[6], reg_vT, R_LEN);

    // ---- RoPE ----
    rope_qk<<<(B_DIM * S_LEN * HEADS * 32) / 256, blk, 0, stream>>>(sum_q, sum_k, S_LEN, 0);
    rope_qk<<<(B_DIM * R_LEN * HEADS * 32) / 256, blk, 0, stream>>>(reg_q, reg_k, R_LEN, S_LEN);

    // ---- Stage 1 ----
    dim3 gA1(S_LEN / 128, HEADS, B_DIM);
    attn<<<gA1, blk, 0, stream>>>(sum_q, sum_k, sum_vT, S_LEN,
                                  reg_k, reg_vT, R_LEN, ssw, srw,
                                  sum_attn, S_LEN);

    // ---- second projections + summary output ----
    gemm_bias<1><<<gS, blk, 0, stream>>>(sum_attn, wbf[8], bsrc[8], k2_bf, S_LEN);
    gemm_bias<2><<<gS, blk, 0, stream>>>(sum_attn, wbf[9], bsrc[9], v2_T,  S_LEN);
    gemm_bias<0><<<gS, blk, 0, stream>>>(sum_attn, wbf[3], bsrc[3], out_sum, S_LEN);

    // ---- Stage 2 ----
    dim3 gA2(R_LEN / 128, HEADS, B_DIM);
    attn<<<gA2, blk, 0, stream>>>(reg_q, k2_bf, v2_T, S_LEN,
                                  reg_k, reg_vT, R_LEN, rsw, rrw,
                                  reg_attn, R_LEN);

    // ---- regular output projection ----
    gemm_bias<0><<<gR, blk, 0, stream>>>(reg_attn, wbf[7], bsrc[7], out_reg, R_LEN);
}
